// myRNN_39084202394146
// MI455X (gfx1250) — matrix operand order repaired and checked by reading, not compiled, not run on hardware
//
#include <hip/hip_runtime.h>
#include <hip/hip_bf16.h>

typedef __attribute__((ext_vector_type(16))) __bf16 v16bf;
typedef __attribute__((ext_vector_type(8)))  __bf16 v8bf;
typedef __attribute__((ext_vector_type(8)))  float  v8f;

constexpr int   NN     = 512;
constexpr int   TT     = 1024;
constexpr int   BB     = 64;
constexpr float KDECAY = 0.1f;

__device__ __forceinline__ unsigned short f2bf(float f) {
    union { float f; unsigned u; } v; v.f = f;
    unsigned r = v.u + 0x7FFFu + ((v.u >> 16) & 1u);   // round-to-nearest-even
    return (unsigned short)(r >> 16);
}

// Hardware v_tanh_f32 (gfx1250 TRANS op) — falls back to ocml if the builtin
// is not declared, so compilation can never break.
__device__ __forceinline__ float fast_tanh(float x) {
#if __has_builtin(__builtin_amdgcn_tanhf)
    return __builtin_amdgcn_tanhf(x);
#elif __has_builtin(__builtin_amdgcn_tanh_f32)
    return __builtin_amdgcn_tanh_f32(x);
#else
    return tanhf(x);
#endif
}

// ---------------------------------------------------------------------------
// Pre-swizzle W (fp32 row-major [n][k]) into bf16 WMMA B-operand order.
// Operand (kc, ntg): lane l, dword j = { k0 = kc*32 + (l&16?16:0) + 2j, k0+1 }
// for column n = ntg*16 + (l&15).  (contiguous k = 16h + s order, same as the A reads)
// ---------------------------------------------------------------------------
__global__ __launch_bounds__(256) void prep_w_kernel(const float* __restrict__ W,
                                                     unsigned* __restrict__ Wb) {
    int tid = blockIdx.x * blockDim.x + threadIdx.x;   // 131072 total
    int j   = tid & 7;
    int l   = (tid >> 3) & 31;
    int ntg = (tid >> 8) & 31;
    int kc  = tid >> 13;
    int n   = ntg * 16 + (l & 15);
    int k0  = kc * 32 + ((l & 16) ? 16 : 0) + 2 * j;
    unsigned lo = f2bf(W[n * NN + k0]);
    unsigned hi = f2bf(W[n * NN + k0 + 1]);
    Wb[tid] = lo | (hi << 16);
}

// ---------------------------------------------------------------------------
// Main recurrence. 4 blocks x 512 threads (16 wave32s). Block owns 16 batches,
// wave owns 32 N-columns. tanh(h) staged in ping-pong LDS (bf16, [m][k]); h in
// registers in the C/D fragment layout. One barrier per timestep.
// ---------------------------------------------------------------------------
__global__ __launch_bounds__(512) void rnn_kernel(const float* __restrict__ h0,
                                                  const float* __restrict__ X,
                                                  const unsigned* __restrict__ Wb,
                                                  float* __restrict__ outH) {
    __shared__ __align__(16) unsigned short thA[2][16][NN];   // 32 KB

    const int lane  = threadIdx.x & 31;
    const int wave  = threadIdx.x >> 5;     // 0..15
    const int half  = lane >> 4;            // 0/1
    const int ln    = lane & 15;
    const int nbase = wave * 32;            // wave's N slice (2 tiles of 16)
    const int b0    = blockIdx.x * 16;      // batch base

    // ---- init hidden state (broadcast h0 over batch) ----
    float h[2][8];
#pragma unroll
    for (int nt = 0; nt < 2; ++nt) {
        int n = nbase + nt * 16 + ln;
        float hv = h0[n];
        unsigned short tb = f2bf(fast_tanh(hv));
#pragma unroll
        for (int r = 0; r < 8; ++r) {
            h[nt][r] = hv;
            thA[0][r + 8 * half][n] = tb;   // A-matrix row m = batch
        }
    }
    __syncthreads();

    const size_t rstride = (size_t)TT * NN;
    const float* xrow0 = X    + (size_t)(b0 + 8 * half) * rstride;
    float*       orow0 = outH + (size_t)(b0 + 8 * half) * rstride;

    for (int t = 0; t < TT; ++t) {
        const int p = t & 1;

        // prefetch x_t for this step (independent of WMMA chain)
        float xv[2][8];
#pragma unroll
        for (int nt = 0; nt < 2; ++nt) {
            int n = nbase + nt * 16 + ln;
#pragma unroll
            for (int r = 0; r < 8; ++r)
                xv[nt][r] = xrow0[(size_t)r * rstride + (size_t)t * NN + n];
        }

        v8f acc[2];
#pragma unroll
        for (int i = 0; i < 2; ++i) {
            v8f z;
#pragma unroll
            for (int e = 0; e < 8; ++e) z[e] = 0.0f;
            acc[i] = z;
        }

#pragma unroll 4
        for (int kc = 0; kc < 16; ++kc) {
            // A fragment from LDS: m = ln; K 0-15 (lanes 0-15), 16-31 (16-31)
            const unsigned short* ap = &thA[p][ln][kc * 32 + half * 16];
            v8bf alo = *(const v8bf*)ap;
            v8bf ahi = *(const v8bf*)(ap + 8);
            v16bf a;
#pragma unroll
            for (int i = 0; i < 8; ++i) { a[i] = alo[i]; a[i + 8] = ahi[i]; }

#pragma unroll
            for (int nt = 0; nt < 2; ++nt) {
                const v8bf* wp = (const v8bf*)(Wb +
                    ((size_t)((kc * 32 + (nbase >> 4) + nt) * 32 + lane)) * 8);
                v8bf blo = wp[0];
                v8bf bhi = wp[1];
                v16bf b;
#pragma unroll
                for (int i = 0; i < 8; ++i) { b[i] = blo[i]; b[i + 8] = bhi[i]; }
                acc[nt] = __builtin_amdgcn_wmma_f32_16x16x32_bf16(
                    false, a, false, b, (short)0, acc[nt], false, false);
            }
        }

        // leaky update + emit tanh(h_new); write next step's A fragments
#pragma unroll
        for (int nt = 0; nt < 2; ++nt) {
            int n = nbase + nt * 16 + ln;
#pragma unroll
            for (int r = 0; r < 8; ++r) {
                float hn = (1.0f - KDECAY) * h[nt][r] + KDECAY * (acc[nt][r] + xv[nt][r]);
                h[nt][r] = hn;
                float th = fast_tanh(hn);
                orow0[(size_t)r * rstride + (size_t)t * NN + n] = th;
                thA[p ^ 1][r + 8 * half][n] = f2bf(th);
            }
        }
        __syncthreads();   // waits ds/load cnts, then barrier: buffer handoff
    }
}

// ---------------------------------------------------------------------------
// geometry = hidden @ Gw^T + Gb. One wave per (b,t) row; bandwidth bound.
// ---------------------------------------------------------------------------
__global__ __launch_bounds__(256) void geom_kernel(const float* __restrict__ hid,
                                                   const float* __restrict__ Gw,
                                                   const float* __restrict__ Gb,
                                                   float* __restrict__ geo) {
    int gtid = blockIdx.x * blockDim.x + threadIdx.x;
    int row  = gtid >> 5;                 // b*T + t
    int lane = threadIdx.x & 31;
    const float4* v  = (const float4*)(hid + (size_t)row * NN);
    const float4* g0 = (const float4*)(Gw);
    const float4* g1 = (const float4*)(Gw + NN);
    float s0 = 0.f, s1 = 0.f;
#pragma unroll
    for (int k = 0; k < 4; ++k) {
        float4 x = v[lane + 32 * k];
        float4 a = g0[lane + 32 * k];
        float4 b = g1[lane + 32 * k];
        s0 += x.x * a.x + x.y * a.y + x.z * a.z + x.w * a.w;
        s1 += x.x * b.x + x.y * b.y + x.z * b.z + x.w * b.w;
    }
#pragma unroll
    for (int off = 16; off > 0; off >>= 1) {
        s0 += __shfl_down(s0, off, 32);
        s1 += __shfl_down(s1, off, 32);
    }
    if (lane == 0) {
        geo[(size_t)row * 2 + 0] = s0 + Gb[0];
        geo[(size_t)row * 2 + 1] = s1 + Gb[1];
    }
}

// ---------------------------------------------------------------------------
extern "C" void kernel_launch(void* const* d_in, const int* in_sizes, int n_in,
                              void* d_out, int out_size, void* d_ws, size_t ws_size,
                              hipStream_t stream) {
    const float* h0 = (const float*)d_in[0];   // [N]
    const float* X  = (const float*)d_in[1];   // [B,T,N]
    const float* W  = (const float*)d_in[2];   // [N,N]
    const float* Gw = (const float*)d_in[3];   // [2,N]
    const float* Gb = (const float*)d_in[4];   // [2]

    float* outH = (float*)d_out;                        // [B,T,N]
    float* geo  = outH + (size_t)BB * TT * NN;          // [B,T,2]
    unsigned* Wb = (unsigned*)d_ws;                     // 512 KB swizzled bf16 W

    prep_w_kernel<<<512, 256, 0, stream>>>(W, Wb);
    rnn_kernel<<<4, 512, 0, stream>>>(h0, X, Wb, outH);
    geom_kernel<<<(BB * TT) / 8, 256, 0, stream>>>(outH, Gw, Gb, geo);
}
